// Net_58033598104011
// MI455X (gfx1250) — matrix operand order repaired and checked by reading, not compiled, not run on hardware
//
#include <hip/hip_runtime.h>
#include <hip/hip_bf16.h>
#include <math.h>

typedef __attribute__((ext_vector_type(16))) _Float16 v16h;
typedef __attribute__((ext_vector_type(8)))  float    v8f;

static constexpr int Nn = 30000, Ee = 400000, C1n = 15000, E2n = 100000,
                     C2n = 7500, Bb = 16;

static inline int cdiv(int a, int b) { return (a + b - 1) / b; }

__device__ __forceinline__ unsigned encf(float f) {
  unsigned u = __float_as_uint(f);
  return (u & 0x80000000u) ? ~u : (u | 0x80000000u);
}
__device__ __forceinline__ float decf(unsigned k) {
  return (k & 0x80000000u) ? __uint_as_float(k ^ 0x80000000u)
                           : __uint_as_float(~k);
}
__device__ __forceinline__ float eluf(float v) {
  return v > 0.f ? v : (expf(v) - 1.f);
}

// B prep: w_b [25, COLS] fp32 -> K-padded col-major f16 [COLS][32]
__global__ void k_prepB(const float* __restrict__ wb, _Float16* __restrict__ B,
                        int cols) {
  int idx = blockIdx.x * blockDim.x + threadIdx.x;
  if (idx >= cols * 32) return;
  int col = idx >> 5, k = idx & 31;
  B[col * 32 + k] = (k < 25) ? (_Float16)wb[k * cols + col] : (_Float16)0.f;
}

__global__ void k_degree(const int* __restrict__ dst, float* __restrict__ cnt,
                         int nE) {
  int e = blockIdx.x * blockDim.x + threadIdx.x;
  if (e < nE) atomicAdd(&cnt[dst[e]], 1.f);
}

// Fused NNConv message kernel: one wave = 16 edges.
// h = relu(ea@wa+ba) built directly into the WMMA A-fragment lane layout;
// per-edge weight tiles via v_wmma_f32_16x16x32_f16 (K 25 padded to 32);
// einsum fold: tile t -> i = t/(CO/16), out group (t%(CO/16))*16;
// scatter-add msg into accum[dst].
template <int CI, int CO>
__global__ __launch_bounds__(256) void k_nnconv_fused(
    const float* __restrict__ eattr, const int* __restrict__ src,
    const int* __restrict__ dst, const float* __restrict__ xin,
    const _Float16* __restrict__ Bws, const float* __restrict__ wa,
    const float* __restrict__ ba, const float* __restrict__ bb,
    float* __restrict__ accum, int nE) {
  constexpr int NG = CO / 16;
  constexpr int NT = (CI * CO) / 16;
  __shared__ float xs[8][16][CI];

  const int wave = threadIdx.x >> 5;
  const int lane = threadIdx.x & 31;
  const int half = lane >> 4;
  const int r    = lane & 15;
  const int ntiles = (nE + 15) >> 4;
  const int tile = blockIdx.x * 8 + wave;
  const bool valid = tile < ntiles;
  const int e0 = valid ? tile * 16 : 0;

  for (int idx = lane; idx < 16 * CI; idx += 32) {
    int m = idx / CI, i = idx % CI;
    int e = e0 + m;
    int s = (e < nE) ? src[e] : 0;
    xs[wave][m][i] = xin[s * CI + i];
  }
  __syncthreads();

  const int erow_a = e0 + r;
  const int ea_i = (erow_a < nE) ? erow_a : 0;
  const float a0 = eattr[ea_i * 3 + 0];
  const float a1 = eattr[ea_i * 3 + 1];
  const float a2 = eattr[ea_i * 3 + 2];
  v16h afrag;
#pragma unroll
  for (int idx = 0; idx < 16; ++idx) {
    int K = half * 16 + idx;
    float h = 0.f;
    if (K < 25) {
      h = ba[K] + a0 * wa[0 * 25 + K] + a1 * wa[1 * 25 + K] +
          a2 * wa[2 * 25 + K];
      h = fmaxf(h, 0.f);
    }
    afrag[idx] = (_Float16)h;
  }

  v8f macc[NG];
#pragma unroll
  for (int g = 0; g < NG; ++g)
#pragma unroll
    for (int q = 0; q < 8; ++q) macc[g][q] = 0.f;

  for (int t = 0; t < NT; ++t) {
    const int i = t / NG;
    const int g = t % NG;
    v16h bfrag = *(const v16h*)(Bws + (size_t)(t * 16 + r) * 32 + half * 16);
    float bv = bb[t * 16 + r];
    v8f c;
#pragma unroll
    for (int q = 0; q < 8; ++q) c[q] = bv;
    c = __builtin_amdgcn_wmma_f32_16x16x32_f16(false, afrag, false, bfrag,
                                               (short)0, c, false, false);
#pragma unroll
    for (int q = 0; q < 8; ++q)
      macc[g][q] += c[q] * xs[wave][half * 8 + q][i];
  }

  if (valid) {
#pragma unroll
    for (int g = 0; g < NG; ++g) {
      const int o = g * 16 + r;
#pragma unroll
      for (int q = 0; q < 8; ++q) {
        const int erow = e0 + half * 8 + q;
        if (erow < nE)
          atomicAdd(&accum[(size_t)dst[erow] * CO + o], macc[g][q]);
      }
    }
  }
}

template <int CI, int CO>
__global__ void k_finalize(const float* __restrict__ xin,
                           const float* __restrict__ accum,
                           const float* __restrict__ cnt,
                           const float* __restrict__ root,
                           const float* __restrict__ bias,
                           float* __restrict__ xout, int nNodes) {
  int idx = blockIdx.x * blockDim.x + threadIdx.x;
  if (idx >= nNodes * CO) return;
  int n = idx / CO, o = idx % CO;
  float v = accum[idx] / fmaxf(cnt[n], 1.f);
#pragma unroll
  for (int i = 0; i < CI; ++i) v += xin[n * CI + i] * root[i * CO + o];
  v += bias[o];
  xout[idx] = eluf(v);
}

__global__ void k_pool_node(const int* __restrict__ cl,
                            const int* __restrict__ batch,
                            const float* __restrict__ pos,
                            float* __restrict__ ncnt, float* __restrict__ psum,
                            int* __restrict__ batp, int nN) {
  int n = blockIdx.x * blockDim.x + threadIdx.x;
  if (n >= nN) return;
  int c = cl[n];
  atomicAdd(&ncnt[c], 1.f);
  atomicAdd(&psum[c * 3 + 0], pos[n * 3 + 0]);
  atomicAdd(&psum[c * 3 + 1], pos[n * 3 + 1]);
  atomicAdd(&psum[c * 3 + 2], pos[n * 3 + 2]);
  atomicMax(&batp[c], batch[n]);
}

__global__ void k_segmax(const float* __restrict__ x, const int* __restrict__ cl,
                         unsigned* __restrict__ keys, int nN, int CO) {
  int idx = blockIdx.x * blockDim.x + threadIdx.x;
  if (idx >= nN * CO) return;
  int n = idx / CO, o = idx % CO;
  atomicMax(&keys[(size_t)cl[n] * CO + o], encf(x[idx]));
}

__global__ void k_decode(const unsigned* __restrict__ keys,
                         const float* __restrict__ cnt, float* __restrict__ out,
                         int nS, int CO) {
  int idx = blockIdx.x * blockDim.x + threadIdx.x;
  if (idx >= nS * CO) return;
  out[idx] = (cnt[idx / CO] > 0.f) ? decf(keys[idx]) : 0.f;
}

__global__ void k_posp(const float* __restrict__ psum,
                       const float* __restrict__ ncnt, float* __restrict__ posp,
                       int nS) {
  int idx = blockIdx.x * blockDim.x + threadIdx.x;
  if (idx >= nS * 3) return;
  posp[idx] = psum[idx] / fmaxf(ncnt[idx / 3], 1.f);
}

__global__ void k_cart(const int* __restrict__ src, const int* __restrict__ dst,
                       const float* __restrict__ posp, float* __restrict__ cart,
                       unsigned* __restrict__ gabs, int nE) {
  int e = blockIdx.x * blockDim.x + threadIdx.x;
  if (e >= nE) return;
  int s = src[e], d = dst[e];
  float m = 0.f;
#pragma unroll
  for (int k = 0; k < 3; ++k) {
    float c = posp[s * 3 + k] - posp[d * 3 + k];
    cart[e * 3 + k] = c;
    m = fmaxf(m, fabsf(c));
  }
  atomicMax(gabs, __float_as_uint(m));
}

__global__ void k_cartnorm(float* __restrict__ cart,
                           const unsigned* __restrict__ gabs, int n) {
  int idx = blockIdx.x * blockDim.x + threadIdx.x;
  if (idx >= n) return;
  float g = __uint_as_float(*gabs);
  cart[idx] = cart[idx] * (0.5f / g) + 0.5f;
}

__global__ void k_c2stats(const int* __restrict__ cl2,
                          const int* __restrict__ batp,
                          float* __restrict__ c2cnt, int* __restrict__ bat2,
                          int nC1) {
  int c = blockIdx.x * blockDim.x + threadIdx.x;
  if (c >= nC1) return;
  int t = cl2[c];
  atomicAdd(&c2cnt[t], 1.f);
  atomicMax(&bat2[t], batp[c]);
}

__global__ void k_gpool(const float* __restrict__ x3,
                        const int* __restrict__ bat2, float* __restrict__ gsum,
                        float* __restrict__ gcnt, int nC2) {
  int idx = blockIdx.x * blockDim.x + threadIdx.x;
  if (idx >= nC2 * 64) return;
  int c = idx / 64, o = idx % 64;
  int b = bat2[c];
  atomicAdd(&gsum[b * 64 + o], x3[idx]);
  if (o == 0) atomicAdd(&gcnt[b], 1.f);
}

__global__ __launch_bounds__(256) void k_head(
    const float* __restrict__ gsum, const float* __restrict__ gcnt,
    const float* __restrict__ fc1w, const float* __restrict__ fc1b,
    const float* __restrict__ fc2w, const float* __restrict__ fc2b,
    float* __restrict__ out) {
  __shared__ float gl[16 * 64];
  __shared__ float hh[16 * 128];
  __shared__ float lg[160];
  __shared__ float rm[16], rl[16];
  int t = threadIdx.x;
  for (int idx = t; idx < 16 * 64; idx += 256)
    gl[idx] = gsum[idx] / fmaxf(gcnt[idx / 64], 1.f);
  __syncthreads();
  for (int idx = t; idx < 16 * 128; idx += 256) {
    int b = idx / 128, j = idx % 128;
    float v = fc1b[j];
    for (int k = 0; k < 64; ++k) v += gl[b * 64 + k] * fc1w[k * 128 + j];
    hh[idx] = eluf(v);
  }
  __syncthreads();
  if (t < 160) {
    int b = t / 10, c = t % 10;
    float v = fc2b[c];
    for (int k = 0; k < 128; ++k) v += hh[b * 128 + k] * fc2w[k * 10 + c];
    lg[t] = v;
  }
  __syncthreads();
  if (t < 16) {
    float m = -1e30f;
    for (int c = 0; c < 10; ++c) m = fmaxf(m, lg[t * 10 + c]);
    float s = 0.f;
    for (int c = 0; c < 10; ++c) s += expf(lg[t * 10 + c] - m);
    rm[t] = m;
    rl[t] = logf(s);
  }
  __syncthreads();
  if (t < 160) out[t] = lg[t] - rm[t / 10] - rl[t / 10];
}

extern "C" void kernel_launch(void* const* d_in, const int* in_sizes, int n_in,
                              void* d_out, int out_size, void* d_ws,
                              size_t ws_size, hipStream_t stream) {
  (void)in_sizes; (void)n_in; (void)out_size; (void)ws_size;
  // setup_inputs() dict order:
  // 0 x, 1 edge_attr, 2 pos, 3 edge_index, 4 batch, 5 cluster1,
  // 6 edge_index2, 7 cluster2, 8 w1a, 9 b1a, 10 w1b, 11 b1b, 12 root1,
  // 13 bias1, 14 w2a, 15 b2a, 16 w2b, 17 b2b, 18 root2, 19 bias2,
  // 20 fc1_w, 21 fc1_b, 22 fc2_w, 23 fc2_b
  const float* x     = (const float*)d_in[0];
  const float* eattr = (const float*)d_in[1];
  const float* pos   = (const float*)d_in[2];
  const int*   ei    = (const int*)d_in[3];
  const int*   batch = (const int*)d_in[4];
  const int*   cl1   = (const int*)d_in[5];
  const int*   ei2   = (const int*)d_in[6];
  const int*   cl2   = (const int*)d_in[7];
  const float* w1a = (const float*)d_in[8],  *b1a = (const float*)d_in[9];
  const float* w1b = (const float*)d_in[10], *b1b = (const float*)d_in[11];
  const float* rt1 = (const float*)d_in[12], *bi1 = (const float*)d_in[13];
  const float* w2a = (const float*)d_in[14], *b2a = (const float*)d_in[15];
  const float* w2b = (const float*)d_in[16], *b2b = (const float*)d_in[17];
  const float* rt2 = (const float*)d_in[18], *bi2 = (const float*)d_in[19];
  const float* f1w = (const float*)d_in[20], *f1b = (const float*)d_in[21];
  const float* f2w = (const float*)d_in[22], *f2b = (const float*)d_in[23];
  float* out = (float*)d_out;
  (void)batch;

  const int *src1 = ei,  *dst1 = ei + Ee;
  const int *src2 = ei2, *dst2 = ei2 + E2n;

  char* ws = (char*)d_ws;
  size_t off = 0;
  auto take = [&](size_t b) -> char* {
    char* p = ws + off;
    off = (off + b + 255) & ~(size_t)255;
    return p;
  };
  // zeroed-every-launch region (single memset)
  float*    ACC1  = (float*)take((size_t)Nn * 32 * 4);
  float*    CNT1  = (float*)take((size_t)Nn * 4);
  unsigned* XPK   = (unsigned*)take((size_t)C1n * 32 * 4);
  float*    NCNT  = (float*)take((size_t)C1n * 4);
  float*    PSUM  = (float*)take((size_t)C1n * 3 * 4);
  int*      BATP  = (int*)take((size_t)C1n * 4);
  unsigned* GABS  = (unsigned*)take(256);
  float*    ACC2  = (float*)take((size_t)C1n * 64 * 4);
  float*    CNT2  = (float*)take((size_t)C1n * 4);
  unsigned* X3K   = (unsigned*)take((size_t)C2n * 64 * 4);
  float*    C2CNT = (float*)take((size_t)C2n * 4);
  int*      BAT2  = (int*)take((size_t)C2n * 4);
  float*    GSUM  = (float*)take((size_t)Bb * 64 * 4);
  float*    GCNT  = (float*)take((size_t)Bb * 4);
  const size_t zbytes = off;
  // fully-overwritten region
  _Float16* BW1  = (_Float16*)take((size_t)32 * 192 * 2);
  _Float16* BW2  = (_Float16*)take((size_t)32 * 2048 * 2);
  float*    X1   = (float*)take((size_t)Nn * 32 * 4);
  float*    XP   = (float*)take((size_t)C1n * 32 * 4);
  float*    POSP = (float*)take((size_t)C1n * 3 * 4);
  float*    CART = (float*)take((size_t)E2n * 3 * 4);
  float*    X2   = (float*)take((size_t)C1n * 64 * 4);
  float*    X3   = (float*)take((size_t)C2n * 64 * 4);

  hipMemsetAsync(d_ws, 0, zbytes, stream);

  k_prepB<<<cdiv(192 * 32, 256), 256, 0, stream>>>(w1b, BW1, 192);
  k_prepB<<<cdiv(2048 * 32, 256), 256, 0, stream>>>(w2b, BW2, 2048);

  // stage 1: NNConv(6->32) over E edges
  k_degree<<<cdiv(Ee, 256), 256, 0, stream>>>(dst1, CNT1, Ee);
  k_nnconv_fused<6, 32><<<cdiv(cdiv(Ee, 16), 8), 256, 0, stream>>>(
      eattr, src1, dst1, x, BW1, w1a, b1a, b1b, ACC1, Ee);
  k_finalize<6, 32><<<cdiv(Nn * 32, 256), 256, 0, stream>>>(x, ACC1, CNT1, rt1,
                                                            bi1, X1, Nn);
  // graclus max-pool to C1
  k_pool_node<<<cdiv(Nn, 256), 256, 0, stream>>>(cl1, (const int*)d_in[4], pos,
                                                 NCNT, PSUM, BATP, Nn);
  k_segmax<<<cdiv(Nn * 32, 256), 256, 0, stream>>>(X1, cl1, XPK, Nn, 32);
  k_decode<<<cdiv(C1n * 32, 256), 256, 0, stream>>>(XPK, NCNT, XP, C1n, 32);
  k_posp<<<cdiv(C1n * 3, 256), 256, 0, stream>>>(PSUM, NCNT, POSP, C1n);

  // normalized Cartesian edge attrs on coarse graph
  k_cart<<<cdiv(E2n, 256), 256, 0, stream>>>(src2, dst2, POSP, CART, GABS, E2n);
  k_cartnorm<<<cdiv(E2n * 3, 256), 256, 0, stream>>>(CART, GABS, E2n * 3);

  // stage 2: NNConv(32->64) over E2 edges
  k_degree<<<cdiv(E2n, 256), 256, 0, stream>>>(dst2, CNT2, E2n);
  k_nnconv_fused<32, 64><<<cdiv(cdiv(E2n, 16), 8), 256, 0, stream>>>(
      CART, src2, dst2, XP, BW2, w2a, b2a, b2b, ACC2, E2n);
  k_finalize<32, 64><<<cdiv(C1n * 64, 256), 256, 0, stream>>>(
      XP, ACC2, CNT2, rt2, bi2, X2, C1n);

  // max_pool_x to C2, global mean pool, FC head + log_softmax
  k_c2stats<<<cdiv(C1n, 256), 256, 0, stream>>>(cl2, BATP, C2CNT, BAT2, C1n);
  k_segmax<<<cdiv(C1n * 64, 256), 256, 0, stream>>>(X2, cl2, X3K, C1n, 64);
  k_decode<<<cdiv(C2n * 64, 256), 256, 0, stream>>>(X3K, C2CNT, X3, C2n, 64);
  k_gpool<<<cdiv(C2n * 64, 256), 256, 0, stream>>>(X3, BAT2, GSUM, GCNT, C2n);
  k_head<<<1, 256, 0, stream>>>(GSUM, GCNT, f1w, f1b, f2w, f2b, out);
}
